// LuluAttention_18330920419553
// MI455X (gfx1250) — matrix operand order repaired and checked by reading, not compiled, not run on hardware
//
#include <hip/hip_runtime.h>
#include <math.h>

#define BB 2
#define SS 2048
#define HH 2048
#define NH 16
#define NKV 4
#define HD 128
#define NROWS (BB * SS) /* 4096 */

typedef __attribute__((ext_vector_type(16))) __bf16 v16bf;
typedef __attribute__((ext_vector_type(8)))  __bf16 v8bf;
typedef __attribute__((ext_vector_type(8)))  float  v8f;

// float -> bf16 bits, round-to-nearest-even
static __device__ __forceinline__ unsigned short f2bf(float f) {
    unsigned u = __builtin_bit_cast(unsigned, f);
    unsigned r = u + 0x7FFFu + ((u >> 16) & 1u);
    return (unsigned short)(r >> 16);
}

// Build a v16bf WMMA fragment from two contiguous 8-element (16B) runs.
static __device__ __forceinline__ v16bf frag2(const unsigned short* p0,
                                              const unsigned short* p1) {
    v8bf lo = *reinterpret_cast<const v8bf*>(p0);
    v8bf hi = *reinterpret_cast<const v8bf*>(p1);
    v16bf r;
#pragma unroll
    for (int i = 0; i < 8; ++i) { r[i] = lo[i]; r[8 + i] = hi[i]; }
    return r;
}

__global__ __launch_bounds__(256) void k_f32_to_bf16(
    const float* __restrict__ in, unsigned short* __restrict__ out, int n) {
    int i = blockIdx.x * blockDim.x + threadIdx.x;
    int stride = gridDim.x * blockDim.x;
    for (; i < n; i += stride) out[i] = f2bf(in[i]);
}

// W:[K][N] f32  ->  WT:[N][K] bf16
__global__ __launch_bounds__(256) void k_transpose_bf16(
    const float* __restrict__ W, unsigned short* __restrict__ WT, int K, int N) {
    long i = (long)blockIdx.x * blockDim.x + threadIdx.x;
    long total = (long)K * N;
    long stride = (long)gridDim.x * blockDim.x;
    for (; i < total; i += stride) {
        int n = (int)(i / K), k = (int)(i % K);
        WT[i] = f2bf(W[(long)k * N + n]);
    }
}

// Per wave: 32x128 output tile (2 row sub-tiles of 16, sharing B fragments).
// MODE 0: rope epilogue, store bf16 [b][head][s][d]   (Q and K)
// MODE 1: plain+bias, store bf16 transposed [b][head][d][s]   (V)
// MODE 2: plain, store f32 row-major [m][n]   (output projection)
template <int MODE>
__global__ __launch_bounds__(128, 1) void k_gemm(
    const unsigned short* __restrict__ A,  // [NROWS][K] bf16
    const unsigned short* __restrict__ BT, // [N][K] bf16
    const float* __restrict__ bias,        // [N] or null
    void* __restrict__ dst, int K, int nHeadsDst) {
    const int lane = threadIdx.x & 31;
    const int wid  = threadIdx.x >> 5;
    const int half = lane >> 4;
    const int l16  = lane & 15;
    const int rbase = blockIdx.x * 128 + wid * 32;  // 4 waves x 32 rows
    const int nbase = blockIdx.y * 128;             // one head's 128 cols

    v8f zero = {};
    v8f acc[2][8];
#pragma unroll
    for (int s = 0; s < 2; ++s)
#pragma unroll
        for (int t = 0; t < 8; ++t) acc[s][t] = zero;

    const unsigned short* arow0 = A + (long)(rbase + l16) * K;
    const unsigned short* arow1 = arow0 + (long)16 * K;
    const unsigned short* bbase = BT + (long)(nbase + l16) * K;

    for (int k0 = 0; k0 < K; k0 += 32) {
        v16bf a0 = frag2(arow0 + k0 + 8 * half, arow0 + k0 + 16 + 8 * half);
        v16bf a1 = frag2(arow1 + k0 + 8 * half, arow1 + k0 + 16 + 8 * half);
        __builtin_prefetch(arow0 + k0 + 32, 0, 3);  // global_prefetch_b8
        __builtin_prefetch(arow1 + k0 + 32, 0, 3);
#pragma unroll
        for (int t = 0; t < 8; ++t) {
            const unsigned short* brow = bbase + (long)16 * t * K + k0;
            v16bf b = frag2(brow + 8 * half, brow + 16 + 8 * half);
            acc[0][t] = __builtin_amdgcn_wmma_f32_16x16x32_bf16(
                false, a0, false, b, (short)0, acc[0][t], false, false);
            acc[1][t] = __builtin_amdgcn_wmma_f32_16x16x32_bf16(
                false, a1, false, b, (short)0, acc[1][t], false, false);
        }
    }

    if (MODE == 2) {
        float* out = (float*)dst;
#pragma unroll
        for (int s = 0; s < 2; ++s)
#pragma unroll
            for (int t = 0; t < 8; ++t) {
                int n = nbase + 16 * t + l16;
#pragma unroll
                for (int r = 0; r < 8; ++r) {
                    int m = rbase + 16 * s + 8 * half + r;
                    out[(long)m * HH + n] = acc[s][t][r];
                }
            }
        return;
    }

#pragma unroll
    for (int t = 0; t < 8; ++t) {
        float bv = bias[nbase + 16 * t + l16];
#pragma unroll
        for (int s = 0; s < 2; ++s)
#pragma unroll
            for (int r = 0; r < 8; ++r) acc[s][t][r] += bv;
    }

    unsigned short* out = (unsigned short*)dst;
    const int head = nbase >> 7;

    if (MODE == 1) {  // V: store transposed [b][head][d][s]
#pragma unroll
        for (int s = 0; s < 2; ++s)
#pragma unroll
            for (int t = 0; t < 8; ++t) {
                int d = 16 * t + l16;
#pragma unroll
                for (int r = 0; r < 8; ++r) {
                    int m = rbase + 16 * s + 8 * half + r;
                    int b = m >> 11, sq = m & (SS - 1);
                    out[((long)(b * nHeadsDst + head) * HD + d) * SS + sq] =
                        f2bf(acc[s][t][r]);
                }
            }
        return;
    }

    // MODE 0: RoPE. Column d pairs with d^64 => tile t pairs with t^4 at the
    // same lane/row in C layout (d = 16*t + l16, 64 = 16*4).
    const float NEG_LN_THETA_64 = -9.2103403719761836f / 64.0f;
#pragma unroll
    for (int s = 0; s < 2; ++s)
#pragma unroll
        for (int t = 0; t < 8; ++t) {
            int d = 16 * t + l16;
            float invf = expf((float)(d & 63) * NEG_LN_THETA_64);
#pragma unroll
            for (int r = 0; r < 8; ++r) {
                int m = rbase + 16 * s + 8 * half + r;
                int b = m >> 11, sq = m & (SS - 1);
                float cs, sn;
                sincosf((float)sq * invf, &sn, &cs);
                float q  = acc[s][t][r];
                float pr = acc[s][t ^ 4][r];
                float rot = (t < 4) ? -pr : pr;
                out[((long)(b * nHeadsDst + head) * SS + sq) * HD + d] =
                    f2bf(q * cs + rot * sn);
            }
        }
}

// Flash attention: block = 4 waves, each wave owns 16 q rows of one (b,h).
__global__ __launch_bounds__(128, 1) void k_attn(
    const unsigned short* __restrict__ Qr,  // [B][NH][S][D]
    const unsigned short* __restrict__ Kr,  // [B][NKV][S][D]
    const unsigned short* __restrict__ Vt,  // [B][NKV][D][S]
    unsigned short* __restrict__ AO) {      // [B*S][NH*D]
    __shared__ __align__(16) unsigned short Plds[4][16][32];
    const int lane = threadIdx.x & 31;
    const int wid  = threadIdx.x >> 5;
    const int half = lane >> 4;
    const int l16  = lane & 15;
    const int qbase = blockIdx.x * 64 + wid * 16;
    const int bh = blockIdx.y;
    const int b = bh >> 4, h = bh & 15;
    const int kv = h >> 2;

    const unsigned short* Qh = Qr + (long)(b * NH + h) * SS * HD;
    const unsigned short* Kh = Kr + (long)(b * NKV + kv) * SS * HD;
    const unsigned short* Vh = Vt + (long)(b * NKV + kv) * HD * SS;

    v16bf qf[4];
    {
        const unsigned short* qrow = Qh + (long)(qbase + l16) * HD;
#pragma unroll
        for (int f = 0; f < 4; ++f)
            qf[f] = frag2(qrow + 32 * f + 8 * half,
                          qrow + 32 * f + 16 + 8 * half);
    }

    v8f zero = {};
    v8f o[8];
#pragma unroll
    for (int t = 0; t < 8; ++t) o[t] = zero;
    float rmax[8], rsum[8];
#pragma unroll
    for (int r = 0; r < 8; ++r) { rmax[r] = -3.0e38f; rsum[r] = 0.0f; }
    const float scale = 0.08838834764831845f;  // 1/sqrt(128)

    for (int k0 = 0; k0 < SS; k0 += 32) {
        __builtin_prefetch(Kh + (long)(k0 + 32 + l16) * HD + 16 * half, 0, 3);
        __builtin_prefetch(Vh + (long)l16 * SS + k0 + 32 + 16 * half, 0, 3);
        v8f sc[2];
#pragma unroll
        for (int nt = 0; nt < 2; ++nt) {
            v8f c = zero;
#pragma unroll
            for (int f = 0; f < 4; ++f) {
                const unsigned short* krow =
                    Kh + (long)(k0 + 16 * nt + l16) * HD + 32 * f;
                v16bf kb = frag2(krow + 8 * half, krow + 16 + 8 * half);
                c = __builtin_amdgcn_wmma_f32_16x16x32_bf16(
                    false, qf[f], false, kb, (short)0, c, false, false);
            }
            sc[nt] = c;
        }
#pragma unroll
        for (int nt = 0; nt < 2; ++nt)
#pragma unroll
            for (int r = 0; r < 8; ++r) sc[nt][r] *= scale;

        // Online softmax per row m = 8*half + r; reduce across the 16-lane half.
        float corr[8];
#pragma unroll
        for (int r = 0; r < 8; ++r) {
            float v = fmaxf(sc[0][r], sc[1][r]);
#pragma unroll
            for (int msk = 1; msk < 16; msk <<= 1)
                v = fmaxf(v, __shfl_xor(v, msk, 32));
            float nm = fmaxf(rmax[r], v);
            corr[r] = expf(rmax[r] - nm);
            rmax[r] = nm;
            float p0 = expf(sc[0][r] - nm);
            float p1 = expf(sc[1][r] - nm);
            Plds[wid][8 * half + r][l16]      = f2bf(p0);
            Plds[wid][8 * half + r][16 + l16] = f2bf(p1);
            float ps = p0 + p1;
#pragma unroll
            for (int msk = 1; msk < 16; msk <<= 1)
                ps += __shfl_xor(ps, msk, 32);
            rsum[r] = rsum[r] * corr[r] + ps;
        }
#pragma unroll
        for (int t = 0; t < 8; ++t)
#pragma unroll
            for (int r = 0; r < 8; ++r) o[t][r] *= corr[r];

        // Reload P from LDS in A-fragment layout (in-order LDS within wave).
        const unsigned short* prow = &Plds[wid][l16][0];
        v16bf pa = frag2(prow + 8 * half, prow + 16 + 8 * half);
#pragma unroll
        for (int t = 0; t < 8; ++t) {
            const unsigned short* vrow =
                Vh + (long)(16 * t + l16) * SS + k0;
            v16bf vb = frag2(vrow + 8 * half, vrow + 16 + 8 * half);
            o[t] = __builtin_amdgcn_wmma_f32_16x16x32_bf16(
                false, pa, false, vb, (short)0, o[t], false, false);
        }
    }

#pragma unroll
    for (int t = 0; t < 8; ++t) {
        int d = 16 * t + l16;
#pragma unroll
        for (int r = 0; r < 8; ++r) {
            int s = qbase + 8 * half + r;
            AO[((long)(b * SS + s)) * (NH * HD) + h * HD + d] =
                f2bf(o[t][r] / rsum[r]);
        }
    }
}

extern "C" void kernel_launch(void* const* d_in, const int* in_sizes, int n_in,
                              void* d_out, int out_size, void* d_ws,
                              size_t ws_size, hipStream_t stream) {
    (void)in_sizes; (void)n_in; (void)out_size; (void)ws_size;
    const float* X  = (const float*)d_in[0];
    const float* Wq = (const float*)d_in[1];
    const float* bq = (const float*)d_in[2];
    const float* Wk = (const float*)d_in[3];
    const float* bk = (const float*)d_in[4];
    const float* Wv = (const float*)d_in[5];
    const float* bv = (const float*)d_in[6];
    const float* Wo = (const float*)d_in[7];
    float* out = (float*)d_out;

    char* ws = (char*)d_ws;
    size_t o = 0;
    unsigned short* Xb  = (unsigned short*)(ws + o); o += (size_t)NROWS * HH * 2;
    unsigned short* WqT = (unsigned short*)(ws + o); o += (size_t)HH * HH * 2;
    unsigned short* WkT = (unsigned short*)(ws + o); o += (size_t)(NKV * HD) * HH * 2;
    unsigned short* WvT = (unsigned short*)(ws + o); o += (size_t)(NKV * HD) * HH * 2;
    unsigned short* WoT = (unsigned short*)(ws + o); o += (size_t)HH * HH * 2;
    unsigned short* Qr  = (unsigned short*)(ws + o); o += (size_t)BB * NH * SS * HD * 2;
    unsigned short* Kr  = (unsigned short*)(ws + o); o += (size_t)BB * NKV * SS * HD * 2;
    unsigned short* Vt  = (unsigned short*)(ws + o); o += (size_t)BB * NKV * HD * SS * 2;
    unsigned short* AO  = (unsigned short*)(ws + o); o += (size_t)NROWS * HH * 2;

    k_f32_to_bf16<<<2048, 256, 0, stream>>>(X, Xb, NROWS * HH);
    k_transpose_bf16<<<2048, 256, 0, stream>>>(Wq, WqT, HH, HH);
    k_transpose_bf16<<<1024, 256, 0, stream>>>(Wk, WkT, HH, NKV * HD);
    k_transpose_bf16<<<1024, 256, 0, stream>>>(Wv, WvT, HH, NKV * HD);
    k_transpose_bf16<<<2048, 256, 0, stream>>>(Wo, WoT, HH, HH);

    dim3 blk(128);
    k_gemm<0><<<dim3(NROWS / 128, (NH * HD) / 128), blk, 0, stream>>>(
        Xb, WqT, bq, Qr, HH, NH);
    k_gemm<0><<<dim3(NROWS / 128, (NKV * HD) / 128), blk, 0, stream>>>(
        Xb, WkT, bk, Kr, HH, NKV);
    k_gemm<1><<<dim3(NROWS / 128, (NKV * HD) / 128), blk, 0, stream>>>(
        Xb, WvT, bv, Vt, HH, NKV);
    k_attn<<<dim3(SS / 64, BB * NH), blk, 0, stream>>>(Qr, Kr, Vt, AO);
    k_gemm<2><<<dim3(NROWS / 128, HH / 128), blk, 0, stream>>>(
        AO, WoT, nullptr, out, HH, 0);
}
